// GromovWassersteinCritic_46901042872765
// MI455X (gfx1250) — matrix operand order repaired and checked by reading, not compiled, not run on hardware
//
#include <hip/hip_runtime.h>
#include <math.h>

// Gromov–Wasserstein critic for MI455X (gfx1250), round 3.
// f16-resident GEMM operands + async global->LDS double-buffered WMMA pipeline.

#define NN 2048
#define DD 256
#define EPS_C 0.1f
#define OUTER_ITERS 5
#define SINK_ITERS 5

typedef __attribute__((ext_vector_type(16))) _Float16 v16h;
typedef __attribute__((ext_vector_type(8)))  _Float16 v8h;
typedef __attribute__((ext_vector_type(8)))  float    v8f;
typedef int v4i __attribute__((vector_size(16)));   // matches builtin param type

#define LDS_AS __attribute__((address_space(3)))

#if defined(__AMDGCN__) && __has_builtin(__builtin_amdgcn_global_load_async_to_lds_b128)
#define GW_HAS_ASYNC 1
#else
#define GW_HAS_ASYNC 0
#endif

__device__ __forceinline__ v16h cat8(v8h a, v8h b) {
  return __builtin_shufflevector(a, b, 0,1,2,3,4,5,6,7,8,9,10,11,12,13,14,15);
}

// One 16-byte async copy global -> LDS.
__device__ __forceinline__ void async_cp16(const _Float16* g, _Float16* l) {
#if GW_HAS_ASYNC
  __builtin_amdgcn_global_load_async_to_lds_b128(
      (v4i*)const_cast<_Float16*>(g), (LDS_AS v4i*)l, 0, 0);
#else
  *reinterpret_cast<v8h*>(l) = *reinterpret_cast<const v8h*>(g);
#endif
}

// Copy this thread's four 16B segments (2 for A, 2 for B) of a 128x32-f16 tile.
__device__ __forceinline__ void copy_tile_seg(const _Float16* ga0, const _Float16* ga1,
                                              const _Float16* gb0, const _Float16* gb1,
                                              _Float16* la0, _Float16* la1,
                                              _Float16* lb0, _Float16* lb1) {
#if GW_HAS_ASYNC
  async_cp16(ga0, la0);
  async_cp16(ga1, la1);
  async_cp16(gb0, lb0);
  async_cp16(gb1, lb1);
#else
  // batched: issue all 4 loads, then all 4 stores (no per-load stall)
  const v8h a0 = *reinterpret_cast<const v8h*>(ga0);
  const v8h a1 = *reinterpret_cast<const v8h*>(ga1);
  const v8h b0 = *reinterpret_cast<const v8h*>(gb0);
  const v8h b1 = *reinterpret_cast<const v8h*>(gb1);
  *reinterpret_cast<v8h*>(la0) = a0;
  *reinterpret_cast<v8h*>(la1) = a1;
  *reinterpret_cast<v8h*>(lb0) = b0;
  *reinterpret_cast<v8h*>(lb1) = b1;
#endif
}

template<int PENDING>
__device__ __forceinline__ void wait_async_tiles() {
#if GW_HAS_ASYNC
#if __has_builtin(__builtin_amdgcn_s_wait_asynccnt)
  __builtin_amdgcn_s_wait_asynccnt(PENDING);
#else
  asm volatile("s_wait_asynccnt %0" :: "i"(PENDING) : "memory");
#endif
#endif
}

// ---------------------------------------------------------------------------
// WMMA GEMM, f16 operands: C = epilogue( A(2048xK) @ Bt^T )
//   A  : row-major f16 (lda), rows are contiguous
//   Bt : row-major f16 (ldb); row j of Bt is column j of the logical B,
//        i.e. we compute A @ Bt^T. Callers pre-arrange data so this is always
//        a contiguous-row copy (C2 fed directly, T fed pre-transposed).
//   EPI=0: f32 C = alpha*acc   EPI=1: f32 C = cc1[i]+cc2[j]+alpha*acc
//   EPI=2: f16 C = (f16)(alpha*acc)
// 128x128x32 block tile, 256 thr = 8 waves (2Mx4N), LDS double-buffered with
// async global->LDS copies overlapping WMMA on the previous tile.
// ---------------------------------------------------------------------------
template<int EPI>
__global__ __launch_bounds__(256)
void wmma_gemm_h(const _Float16* __restrict__ A, int lda,
                 const _Float16* __restrict__ Bt, int ldb,
                 void* __restrict__ Cout, int ldc, int Kdim, float alpha,
                 const float* __restrict__ cc1, const float* __restrict__ cc2)
{
  constexpr int LS = 40;   // LDS row stride in halves (80B rows, b128-aligned)
  __shared__ __align__(16) _Float16 lA[2][128 * LS];
  __shared__ __align__(16) _Float16 lB[2][128 * LS];

  const int tid  = threadIdx.x;
  const int lane = tid & 31;
  const int wv   = tid >> 5;
  const int wM   = wv >> 2;                 // 0..1 -> 64 rows
  const int wN   = wv & 3;                  // 0..3 -> 32 cols
  const int l    = lane & 15;
  const int hi   = lane >> 4;
  const int bm   = blockIdx.y * 128;
  const int bn   = blockIdx.x * 128;

  const int r0  = tid >> 2;                 // 0..63 (this thread's tile row)
  const int seg = (tid & 3) * 8;            // 16B segment within a 32-half row

  const _Float16* gA = A  + (size_t)(bm + r0) * lda + seg;
  const _Float16* gB = Bt + (size_t)(bn + r0) * ldb + seg;
  const size_t a64 = (size_t)64 * lda, b64 = (size_t)64 * ldb;
  const int lo0 = r0 * LS + seg, lo1 = (r0 + 64) * LS + seg;

  v8f acc[4][2];
#pragma unroll
  for (int a = 0; a < 4; ++a)
#pragma unroll
    for (int b = 0; b < 2; ++b) { v8f z = {}; acc[a][b] = z; }

  // prologue: stage tile 0 into buffer 0
  copy_tile_seg(gA, gA + a64, gB, gB + b64,
                &lA[0][lo0], &lA[0][lo1], &lB[0][lo0], &lB[0][lo1]);

  const int nT = Kdim / 32;
  for (int it = 0; it < nT; ++it) {
    const int cur = it & 1;
    if (it + 1 < nT) {
      const int kn = (it + 1) * 32;
      copy_tile_seg(gA + kn, gA + kn + a64, gB + kn, gB + kn + b64,
                    &lA[cur ^ 1][lo0], &lA[cur ^ 1][lo1],
                    &lB[cur ^ 1][lo0], &lB[cur ^ 1][lo1]);
      wait_async_tiles<4>();                // tile `it` landed; next tile in flight
    } else {
      wait_async_tiles<0>();
    }
    __syncthreads();

    // gather fragments per CDNA5 16-bit WMMA layouts
    v16h aF[4], bF[2];
#pragma unroll
    for (int wm = 0; wm < 4; ++wm) {
      const int row = wM * 64 + wm * 16 + l;
      const int off = row * LS + hi * 8;    // lo lanes K0..7, hi lanes K8..15
      v8h p0 = *reinterpret_cast<const v8h*>(&lA[cur][off]);
      v8h p1 = *reinterpret_cast<const v8h*>(&lA[cur][off + 16]);
      aF[wm] = cat8(p0, p1);
    }
#pragma unroll
    for (int wn = 0; wn < 2; ++wn) {
      const int col = wN * 32 + wn * 16 + l;
      const int off = col * LS + hi * 8;    // lo lanes K0..7, hi lanes K8..15
      v8h p0 = *reinterpret_cast<const v8h*>(&lB[cur][off]);
      v8h p1 = *reinterpret_cast<const v8h*>(&lB[cur][off + 16]);
      bF[wn] = cat8(p0, p1);
    }
#pragma unroll
    for (int wm = 0; wm < 4; ++wm)
#pragma unroll
      for (int wn = 0; wn < 2; ++wn)
        acc[wm][wn] = __builtin_amdgcn_wmma_f32_16x16x32_f16(
            false, aF[wm], false, bF[wn], (short)0, acc[wm][wn], false, false);
    __syncthreads();                        // release `cur` before next writes
  }

  // epilogue: lane l = column, VGPR r = row (hi*8 + r)
#pragma unroll
  for (int wm = 0; wm < 4; ++wm)
#pragma unroll
    for (int wn = 0; wn < 2; ++wn) {
      const int gr = bm + wM * 64 + wm * 16 + hi * 8;
      const int gc = bn + wN * 32 + wn * 16 + l;
#pragma unroll
      for (int r = 0; r < 8; ++r) {
        const float v = alpha * acc[wm][wn][r];
        const size_t o = (size_t)(gr + r) * ldc + gc;
        if (EPI == 0)      ((float*)Cout)[o] = v;
        else if (EPI == 1) ((float*)Cout)[o] = cc1[gr + r] + cc2[gc] + v;
        else               ((_Float16*)Cout)[o] = (_Float16)v;
      }
    }
}

// ---------------------------------------------------------------------------
// Helper kernels
// ---------------------------------------------------------------------------
__global__ void fill_kernel(float* __restrict__ p, float v, size_t cnt) {
  for (size_t i = (size_t)blockIdx.x * blockDim.x + threadIdx.x; i < cnt;
       i += (size_t)gridDim.x * blockDim.x) p[i] = v;
}

__global__ void fillh_kernel(_Float16* __restrict__ p, float v, size_t cnt) {
  const _Float16 h = (_Float16)v;
  for (size_t i = (size_t)blockIdx.x * blockDim.x + threadIdx.x; i < cnt;
       i += (size_t)gridDim.x * blockDim.x) p[i] = h;
}

__global__ void tohalf_kernel(const float* __restrict__ src, _Float16* __restrict__ dst,
                              size_t cnt) {
  for (size_t i = (size_t)blockIdx.x * blockDim.x + threadIdx.x; i < cnt;
       i += (size_t)gridDim.x * blockDim.x) dst[i] = (_Float16)src[i];
}

__global__ void rownorm_kernel(const float* __restrict__ x,
                               float* __restrict__ sq1, float* __restrict__ sq2) {
  __shared__ float red[DD];
  const int tid = threadIdx.x, row = blockIdx.x;
  const float v = x[(size_t)row * DD + tid];
  red[tid] = v * v; __syncthreads();
  for (int st = DD / 2; st > 0; st >>= 1) {
    if (tid < st) red[tid] += red[tid + st];
    __syncthreads();
  }
  if (tid == 0) { if (row < NN) sq1[row] = red[0]; else sq2[row - NN] = red[0]; }
}

// In-place: SC holds the Gram matrix, becomes raw squared distances; max -> bits
__global__ void formC_kernel(float* __restrict__ SC, const float* __restrict__ sq,
                             unsigned* __restrict__ maxbits) {
  __shared__ float red[256];
  const int tid = threadIdx.x;
  const size_t total = (size_t)NN * NN;
  float lmax = 0.f;
  for (size_t idx = (size_t)blockIdx.x * 256 + tid; idx < total;
       idx += (size_t)gridDim.x * 256) {
    const int i = (int)(idx >> 11), j = (int)(idx & (NN - 1));
    const float d = fmaxf(sq[i] + sq[j] - 2.f * SC[idx], 0.f);
    SC[idx] = d;
    lmax = fmaxf(lmax, d);
  }
  red[tid] = lmax; __syncthreads();
  for (int st = 128; st > 0; st >>= 1) {
    if (tid < st) red[tid] = fmaxf(red[tid], red[tid + st]);
    __syncthreads();
  }
  if (tid == 0) atomicMax(maxbits, __float_as_uint(red[0]));  // values >= 0
}

// Ch = (f16)(Craw / max)
__global__ void scaleh_kernel(const float* __restrict__ src, _Float16* __restrict__ dst,
                              const unsigned* __restrict__ maxbits) {
  const float inv = 1.f / __uint_as_float(*maxbits);
  const size_t total = (size_t)NN * NN;
  for (size_t idx = (size_t)blockIdx.x * blockDim.x + threadIdx.x; idx < total;
       idx += (size_t)gridDim.x * blockDim.x) dst[idx] = (_Float16)(src[idx] * inv);
}

// cc[i] = (1/NN) * sum_j Ch[i][j]^2   ((C*C) @ p with uniform p)
__global__ void rowsq_kernel(const _Float16* __restrict__ Ch, float* __restrict__ cc) {
  __shared__ float red[256];
  const int tid = threadIdx.x, row = blockIdx.x;
  float s = 0.f;
  for (int j = tid; j < NN; j += 256) {
    const float v = (float)Ch[(size_t)row * NN + j];
    s += v * v;
  }
  red[tid] = s; __syncthreads();
  for (int st = 128; st > 0; st >>= 1) {
    if (tid < st) red[tid] += red[tid + st];
    __syncthreads();
  }
  if (tid == 0) cc[row] = red[0] * (1.0f / NN);
}

// g[j] = eps*(logq - LSE_i((f[i]-M[i,j])/eps)) : 64 cols/block, 4-way row split
__global__ void col_lse_kernel(const float* __restrict__ Mm, const float* __restrict__ f,
                               float* __restrict__ g, float eps, float logq) {
  __shared__ float smx[256], ssm[256];
  const int tid = threadIdx.x, cr = tid & 63, rr = tid >> 6;
  const int j = blockIdx.x * 64 + cr;
  const float inv = 1.f / eps;
  float mx = -INFINITY, sm = 0.f;
  for (int i = rr; i < NN; i += 4) {
    const float v = (f[i] - Mm[(size_t)i * NN + j]) * inv;
    if (v > mx) { sm = sm * __expf(mx - v) + 1.f; mx = v; }
    else        { sm += __expf(v - mx); }
  }
  smx[tid] = mx; ssm[tid] = sm; __syncthreads();
  if (rr == 0) {
    float m0 = smx[tid], s0 = ssm[tid];
    for (int k = 1; k < 4; ++k) {
      const float m1 = smx[tid + 64 * k], s1 = ssm[tid + 64 * k];
      if (m1 > m0) { s0 = s0 * __expf(m0 - m1) + s1; m0 = m1; }
      else         { s0 += s1 * __expf(m1 - m0); }
    }
    g[j] = eps * (logq - (m0 + __logf(s0)));
  }
}

// f[i] = eps*(logp - LSE_j((g[j]-M[i,j])/eps)) : one block per row
__global__ void row_lse_kernel(const float* __restrict__ Mm, const float* __restrict__ g,
                               float* __restrict__ f, float eps, float logp) {
  __shared__ float smx[256], ssm[256];
  const int tid = threadIdx.x, i = blockIdx.x;
  const float inv = 1.f / eps;
  float mx = -INFINITY, sm = 0.f;
  for (int j = tid; j < NN; j += 256) {
    const float v = (g[j] - Mm[(size_t)i * NN + j]) * inv;
    if (v > mx) { sm = sm * __expf(mx - v) + 1.f; mx = v; }
    else        { sm += __expf(v - mx); }
  }
  smx[tid] = mx; ssm[tid] = sm; __syncthreads();
  for (int st = 128; st > 0; st >>= 1) {
    if (tid < st) {
      float m0 = smx[tid], s0 = ssm[tid];
      const float m1 = smx[tid + st], s1 = ssm[tid + st];
      if (m1 > m0) { s0 = s0 * __expf(m0 - m1) + s1; m0 = m1; }
      else         { s0 += s1 * __expf(m1 - m0); }
      smx[tid] = m0; ssm[tid] = s0;
    }
    __syncthreads();
  }
  if (tid == 0) f[i] = eps * (logp - (smx[0] + __logf(ssm[0])));
}

// T f32 (row-major) and Tht = (nm*T)^T in f16, via 64x64 LDS tile transpose
__global__ void updateT_kernel(float* __restrict__ T, _Float16* __restrict__ Tht,
                               const float* __restrict__ Mm,
                               const float* __restrict__ f, const float* __restrict__ g,
                               float inveps, float nm) {
  __shared__ _Float16 tile[64][65];
  const int tid = threadIdx.x;
  const int i0 = blockIdx.y * 64, j0 = blockIdx.x * 64;
  const int c = tid & 63, rb = tid >> 6;
#pragma unroll 4
  for (int p = 0; p < 16; ++p) {
    const int r = rb + p * 4;
    const size_t idx = (size_t)(i0 + r) * NN + (j0 + c);
    const float e = __expf((f[i0 + r] + g[j0 + c] - Mm[idx]) * inveps);
    T[idx] = e;
    tile[r][c] = (_Float16)(e * nm);
  }
  __syncthreads();
#pragma unroll 4
  for (int p = 0; p < 16; ++p) {
    const int r = rb + p * 4;                       // row of transposed tile
    Tht[(size_t)(j0 + r) * NN + (i0 + c)] = tile[c][r];
  }
}

__global__ void dot_kernel(const float* __restrict__ Mm, const float* __restrict__ T,
                           float* __restrict__ out) {
  __shared__ float red[256];
  const int tid = threadIdx.x;
  const size_t total = (size_t)NN * NN;
  float s = 0.f;
  for (size_t idx = (size_t)blockIdx.x * 256 + tid; idx < total;
       idx += (size_t)gridDim.x * 256) s += Mm[idx] * T[idx];
  red[tid] = s; __syncthreads();
  for (int st = 128; st > 0; st >>= 1) {
    if (tid < st) red[tid] += red[tid + st];
    __syncthreads();
  }
  if (tid == 0) atomicAdd(out, red[0]);
}

// ---------------------------------------------------------------------------
// Orchestration. Workspace: Mm, T (f32, 16MB each); C1h, C2h, Tht, Wh (f16,
// 8MB each); xh (f16, 2MB); 6 NN-float vectors + 2 uints.  ~66.1 MB total.
// ---------------------------------------------------------------------------
extern "C" void kernel_launch(void* const* d_in, const int* in_sizes, int n_in,
                              void* d_out, int out_size, void* d_ws, size_t ws_size,
                              hipStream_t stream) {
  (void)in_sizes; (void)n_in; (void)out_size; (void)ws_size;
  const float* x = (const float*)d_in[0];   // (4096, 256) f32
  // d_in[1] labels: [0]*2048 + [1]*2048 -> stable argsort is identity; unused.
  float* out = (float*)d_out;

  const size_t mat = (size_t)NN * NN;
  float* Mm = (float*)d_ws;
  float* T  = Mm + mat;
  _Float16* C1h = (_Float16*)(T + mat);
  _Float16* C2h = C1h + mat;
  _Float16* Tht = C2h + mat;
  _Float16* Wh  = Tht + mat;
  _Float16* xh  = Wh + mat;                 // 4096*256 halves
  float* sq1 = (float*)(xh + (size_t)2 * NN * DD);
  float* sq2 = sq1 + NN;
  float* cc1 = sq2 + NN;
  float* cc2 = cc1 + NN;
  float* f   = cc2 + NN;
  float* g   = f + NN;
  unsigned* maxb = (unsigned*)(g + NN);

  const float eps  = EPS_C;
  const float logp = -logf((float)NN);
  const float logq = -logf((float)NN);
  const float nm   = (float)NN * (float)NN;
  const dim3 gg(NN / 128, NN / 128);

  fill_kernel<<<1, 64, 0, stream>>>((float*)maxb, 0.f, 2);
  rownorm_kernel<<<2 * NN, DD, 0, stream>>>(x, sq1, sq2);
  tohalf_kernel<<<1024, 256, 0, stream>>>(x, xh, (size_t)2 * NN * DD);

  // C1h = (f16) relu(sq1_i + sq1_j - 2 x1 x1^T) / max
  wmma_gemm_h<0><<<gg, 256, 0, stream>>>(xh, DD, xh, DD, Mm, NN, DD, 1.f,
                                         nullptr, nullptr);
  formC_kernel<<<1024, 256, 0, stream>>>(Mm, sq1, maxb);
  scaleh_kernel<<<1024, 256, 0, stream>>>(Mm, C1h, maxb);

  const _Float16* x2h = xh + (size_t)NN * DD;
  wmma_gemm_h<0><<<gg, 256, 0, stream>>>(x2h, DD, x2h, DD, Mm, NN, DD, 1.f,
                                         nullptr, nullptr);
  formC_kernel<<<1024, 256, 0, stream>>>(Mm, sq2, maxb + 1);
  scaleh_kernel<<<1024, 256, 0, stream>>>(Mm, C2h, maxb + 1);

  rowsq_kernel<<<NN, 256, 0, stream>>>(C1h, cc1);
  rowsq_kernel<<<NN, 256, 0, stream>>>(C2h, cc2);

  fill_kernel<<<2048, 256, 0, stream>>>(T, 1.f / nm, mat);   // T0 = p q^T
  fillh_kernel<<<2048, 256, 0, stream>>>(Tht, 1.f, mat);     // nm*T0 = 1

  for (int ot = 0; ot <= OUTER_ITERS; ++ot) {
    // Wh = C1 @ Ts          (Ts = nm*T, fed pre-transposed: Bt = Tht)
    wmma_gemm_h<2><<<gg, 256, 0, stream>>>(C1h, NN, Tht, NN, Wh, NN, NN, 1.f,
                                           nullptr, nullptr);
    // Mm = cc1_i + cc2_j - (2/nm) * W @ C2^T   (Bt = C2 rows)
    wmma_gemm_h<1><<<gg, 256, 0, stream>>>(Wh, NN, C2h, NN, Mm, NN, NN,
                                           -2.f / nm, cc1, cc2);
    if (ot < OUTER_ITERS) {
      fill_kernel<<<8, 256, 0, stream>>>(f, 0.f, NN);
      fill_kernel<<<8, 256, 0, stream>>>(g, 0.f, NN);
      for (int s = 0; s < SINK_ITERS; ++s) {
        col_lse_kernel<<<NN / 64, 256, 0, stream>>>(Mm, f, g, eps, logq);
        row_lse_kernel<<<NN, 256, 0, stream>>>(Mm, g, f, eps, logp);
      }
      updateT_kernel<<<dim3(NN / 64, NN / 64), 256, 0, stream>>>(
          T, Tht, Mm, f, g, 1.f / eps, nm);
    } else {
      fill_kernel<<<1, 32, 0, stream>>>(out, 0.f, 1);
      dot_kernel<<<1024, 256, 0, stream>>>(Mm, T, out);
    }
  }
}
